// GATBlock_26130581029644
// MI455X (gfx1250) — matrix operand order repaired and checked by reading, not compiled, not run on hardware
//
#include <hip/hip_runtime.h>
#include <math.h>

typedef __attribute__((ext_vector_type(16))) _Float16 v16h;
typedef __attribute__((ext_vector_type(8)))  _Float16 v8h;
typedef __attribute__((ext_vector_type(4)))  _Float16 v4h;
typedef __attribute__((ext_vector_type(8)))  float    v8f;
typedef __attribute__((ext_vector_type(4)))  float    v4f;

#define C_DIM 128
#define NEG_SLOPE 0.01f
#define LN_EPS 1e-5f
#define NCHUNK 4
#define TILE_J 128

// -------------------- reduction helper --------------------
__device__ __forceinline__ float block_reduce_sum(float v, float* red, int tid, int nthr) {
  red[tid] = v; __syncthreads();
  for (int s = nthr >> 1; s > 0; s >>= 1) {
    if (tid < s) red[tid] += red[tid + s];
    __syncthreads();
  }
  float r = red[0]; __syncthreads();
  return r;
}

// -------------------- WMMA fragment helpers --------------------
// A: 16x32 f16 row-major (stride lda). lane 0-15: M=lane, K {0..7,16..23};
// lane 16-31: M=lane-16, K {8..15,24..31}.  Two contiguous 8-half runs -> 2x B128.
__device__ __forceinline__ v16h load_frag_a(const _Float16* p, int lda, int lane) {
  int m = lane & 15, g = lane >> 4;
  const _Float16* base = p + (size_t)m * lda + g * 8;
  v8h lo = *(const v8h*)(base);
  v8h hi = *(const v8h*)(base + 16);
  return __builtin_shufflevector(lo, hi, 0, 1, 2, 3, 4, 5, 6, 7, 8, 9, 10, 11, 12, 13, 14, 15);
}
// B: 32x16 (KxN) from K-major storage: element [n][k] at p[n*ldt + k].
// lanes 0-15: N=lane, K {0..7,16..23}; lanes 16-31: N=lane-16, K {8..15,24..31}. 2x B128 per lane.
__device__ __forceinline__ v16h load_frag_b_kmajor(const _Float16* p, size_t ldt, int lane) {
  int nn = lane & 15, g = lane >> 4;
  const _Float16* base = p + (size_t)nn * ldt + g * 8;
  v8h lo = *(const v8h*)(base);
  v8h hi = *(const v8h*)(base + 16);
  return __builtin_shufflevector(lo, hi, 0, 1, 2, 3, 4, 5, 6, 7, 8, 9, 10, 11, 12, 13, 14, 15);
}

// -------------------- kernels --------------------
__global__ void k_ln1(const float* __restrict__ x, const float* __restrict__ gg,
                      const float* __restrict__ bb, _Float16* __restrict__ xn16) {
  __shared__ float red[C_DIM];
  int r = blockIdx.x, t = threadIdx.x;
  float v = x[(size_t)r * C_DIM + t];
  float mu = block_reduce_sum(v, red, t, C_DIM) * (1.f / C_DIM);
  float d = v - mu;
  float var = block_reduce_sum(d * d, red, t, C_DIM) * (1.f / C_DIM);
  float y = d * rsqrtf(var + LN_EPS) * gg[t] + bb[t];
  xn16[(size_t)r * C_DIM + t] = (_Float16)y;
}

// transpose-convert weights: dst[k][n][c] = src[k][c][n]  (K-major B operand)
__global__ void k_cvt_t(const float* __restrict__ src, _Float16* __restrict__ dst, int total) {
  int i = blockIdx.x * blockDim.x + threadIdx.x;
  if (i >= total) return;
  int k = i / (C_DIM * C_DIM);
  int rem = i % (C_DIM * C_DIM);
  int r = rem / C_DIM, c = rem % C_DIM;
  dst[(size_t)k * C_DIM * C_DIM + (size_t)c * C_DIM + r] = (_Float16)src[i];
}

// h[k] = x_norm @ W0[k]; outputs: row-major h16 (for scores) + K-major ht16 (for attn GEMM)
__global__ void k_gemm_h(const _Float16* __restrict__ xn16, const _Float16* __restrict__ w0t,
                         _Float16* __restrict__ h16, _Float16* __restrict__ ht16, int n) {
  int lane = threadIdx.x & 31;
  int wg = blockIdx.x * (blockDim.x >> 5) + (threadIdx.x >> 5);
  int tiles_per_head = (n / 16) * (C_DIM / 16);
  if (wg >= 2 * tiles_per_head) return;
  int head = wg / tiles_per_head;
  int rem = wg % tiles_per_head;
  int mt = rem / (C_DIM / 16), nt = rem % (C_DIM / 16);
  const _Float16* A = xn16 + (size_t)(mt * 16) * C_DIM;
  const _Float16* B = w0t + (size_t)head * C_DIM * C_DIM + (size_t)(nt * 16) * C_DIM;
  v8f c = {};
#pragma unroll
  for (int kk = 0; kk < C_DIM; kk += 32) {
    v16h a = load_frag_a(A + kk, C_DIM, lane);
    v16h b = load_frag_b_kmajor(B + kk, C_DIM, lane);
    c = __builtin_amdgcn_wmma_f32_16x16x32_f16(false, a, false, b, (short)0, c, false, false);
  }
  int nn = lane & 15, g = lane >> 4;
  // row-major store (scalar, small)
  _Float16* D = h16 + (size_t)head * n * C_DIM + (size_t)(mt * 16) * C_DIM + nt * 16;
#pragma unroll
  for (int v = 0; v < 8; ++v) D[(g * 8 + v) * C_DIM + nn] = (_Float16)c[v];
  // K-major (node-contiguous) store: one packed 16B store
  v8h hv;
#pragma unroll
  for (int v = 0; v < 8; ++v) hv[v] = (_Float16)c[v];
  _Float16* T = ht16 + (size_t)head * C_DIM * n + (size_t)(nt * 16 + nn) * n + mt * 16 + g * 8;
  *(v8h*)T = hv;
}

// er[k][i] = h[k][i]·a_l ; ec[k][i] = h[k][i]·a_r
__global__ void k_scores(const _Float16* __restrict__ h16, const float* __restrict__ a,
                         float* __restrict__ er, float* __restrict__ ec, int n) {
  __shared__ float red[C_DIM];
  int r = blockIdx.x, t = threadIdx.x;
  for (int k = 0; k < 2; ++k) {
    float hv = (float)h16[(size_t)k * n * C_DIM + (size_t)r * C_DIM + t];
    float dl = block_reduce_sum(hv * a[k * 2 * C_DIM + t], red, t, C_DIM);
    float dr = block_reduce_sum(hv * a[k * 2 * C_DIM + C_DIM + t], red, t, C_DIM);
    if (t == 0) { er[(size_t)k * n + r] = dl; ec[(size_t)k * n + r] = dr; }
  }
}

__device__ __forceinline__ float lrelu(float e) { return e > 0.f ? e : NEG_SLOPE * e; }

// per-row softmax stats (max + 1/sum), both heads, float4-streamed adj
__global__ void k_stats(const float* __restrict__ adj, const float* __restrict__ er,
                        const float* __restrict__ ec, float* __restrict__ mrow,
                        float* __restrict__ sinv, int n) {
  __shared__ float red[256];
  int i = blockIdx.x, t = threadIdx.x;
  float er0 = er[i], er1 = er[n + i];
  float m0 = -3.0e38f, m1 = -3.0e38f;
  for (int j = t * 4; j < n; j += 1024) {
    v4f ad = *(const v4f*)(adj + (size_t)i * n + j);
    v4f c0 = *(const v4f*)(ec + j);
    v4f c1 = *(const v4f*)(ec + n + j);
#pragma unroll
    for (int q = 0; q < 4; ++q) {
      if (ad[q] > 0.f || (j + q) == i) {
        m0 = fmaxf(m0, lrelu(er0 + c0[q]));
        m1 = fmaxf(m1, lrelu(er1 + c1[q]));
      }
    }
  }
  red[t] = m0; __syncthreads();
  for (int s = 128; s > 0; s >>= 1) { if (t < s) red[t] = fmaxf(red[t], red[t + s]); __syncthreads(); }
  m0 = red[0]; __syncthreads();
  red[t] = m1; __syncthreads();
  for (int s = 128; s > 0; s >>= 1) { if (t < s) red[t] = fmaxf(red[t], red[t + s]); __syncthreads(); }
  m1 = red[0]; __syncthreads();
  float s0 = 0.f, s1 = 0.f;
  for (int j = t * 4; j < n; j += 1024) {
    v4f ad = *(const v4f*)(adj + (size_t)i * n + j);
    v4f c0 = *(const v4f*)(ec + j);
    v4f c1 = *(const v4f*)(ec + n + j);
#pragma unroll
    for (int q = 0; q < 4; ++q) {
      if (ad[q] > 0.f || (j + q) == i) {
        s0 += __expf(lrelu(er0 + c0[q]) - m0);
        s1 += __expf(lrelu(er1 + c1[q]) - m1);
      }
    }
  }
  s0 = block_reduce_sum(s0, red, t, 256);
  s1 = block_reduce_sum(s1, red, t, 256);
  if (t == 0) {
    mrow[i] = m0; mrow[n + i] = m1;
    sinv[i] = 1.f / s0; sinv[n + i] = 1.f / s1;
  }
}

// fused attention pass: 16-row block x 128-col j-tiles, LDS double-buffered.
// phase1 (all 256 thr): atten tile both heads -> NT atten_agg store + LDS f16.
// phase2 (8 waves): 4 k-steps x 2 heads of WMMA against K-major h. One barrier/iter.
__global__ void k_attn(const float* __restrict__ adj, const float* __restrict__ er,
                       const float* __restrict__ ec, const float* __restrict__ mrow,
                       const float* __restrict__ sinv, const _Float16* __restrict__ ht16,
                       float* __restrict__ agg, float* __restrict__ part, int n) {
  __shared__ __align__(16) _Float16 la[2][2][16][TILE_J];   // [buf][head][row][col]
  __shared__ float ler[2][16], lm[2][16], lsi[2][16];
  int t = threadIdx.x;
  int lane = t & 31, wv = t >> 5;
  int i0 = blockIdx.x * 16;
  int chunk = blockIdx.y;
  int jbeg = chunk * (n / NCHUNK), jend = jbeg + n / NCHUNK;
  if (t < 32) {
    int k = t >> 4, ii = t & 15;
    ler[k][ii] = er[(size_t)k * n + i0 + ii];
    lm[k][ii]  = mrow[(size_t)k * n + i0 + ii];
    lsi[k][ii] = sinv[(size_t)k * n + i0 + ii];
  }
  __syncthreads();
  v8f c0 = {}, c1 = {};
  int n0 = wv * 16;
  const _Float16* HT0 = ht16 + (size_t)n0 * n;                          // head 0, feature-major
  const _Float16* HT1 = ht16 + (size_t)C_DIM * n + (size_t)n0 * n;      // head 1
  int p_ii = t >> 4;            // 0..15
  int p_jj = (t & 15) * 4;      // 0..60 (plus +64 sub-tile)
  const float er0 = ler[0][p_ii], er1 = ler[1][p_ii];
  const float mm0 = lm[0][p_ii],  mm1 = lm[1][p_ii];
  const float ss0 = lsi[0][p_ii], ss1 = lsi[1][p_ii];
  const float* adjRow = adj + (size_t)(i0 + p_ii) * n;
  float* aggRow = agg + (size_t)(i0 + p_ii) * n;
  int buf = 0;
  for (int j0 = jbeg; j0 < jend; j0 += TILE_J) {
    // ---- phase 1: two float4 sub-tiles per thread, both heads ----
#pragma unroll
    for (int half = 0; half < 2; ++half) {
      int jj = p_jj + half * 64;
      int j = j0 + jj;
      v4f ad = *(const v4f*)(adjRow + j);
      __builtin_prefetch(adjRow + j + TILE_J, 0, 1);   // next tile of adj (speculative)
      v4f e0 = *(const v4f*)(ec + j);
      v4f e1 = *(const v4f*)(ec + n + j);
      float w0[4], w1[4];
#pragma unroll
      for (int q = 0; q < 4; ++q) {
        bool msk = (ad[q] > 0.f) || ((j + q) == (i0 + p_ii));
        w0[q] = msk ? __expf(lrelu(er0 + e0[q]) - mm0) * ss0 : 0.f;
        w1[q] = msk ? __expf(lrelu(er1 + e1[q]) - mm1) * ss1 : 0.f;
      }
      v4h pk0 = {(_Float16)w0[0], (_Float16)w0[1], (_Float16)w0[2], (_Float16)w0[3]};
      v4h pk1 = {(_Float16)w1[0], (_Float16)w1[1], (_Float16)w1[2], (_Float16)w1[3]};
      *(v4h*)&la[buf][0][p_ii][jj] = pk0;
      *(v4h*)&la[buf][1][p_ii][jj] = pk1;
      v4f ag = {0.5f * (w0[0] + w1[0]), 0.5f * (w0[1] + w1[1]),
                0.5f * (w0[2] + w1[2]), 0.5f * (w0[3] + w1[3])};
      __builtin_nontemporal_store(ag, (v4f*)(aggRow + j));   // write-once stream: keep L2 clean
    }
    __syncthreads();
    // ---- phase 2: 4 k-steps x 2 heads of WMMA ----
#pragma unroll
    for (int ks = 0; ks < TILE_J; ks += 32) {
      v16h a0 = load_frag_a(&la[buf][0][0][ks], TILE_J, lane);
      v16h a1 = load_frag_a(&la[buf][1][0][ks], TILE_J, lane);
      v16h b0 = load_frag_b_kmajor(HT0 + j0 + ks, n, lane);
      v16h b1 = load_frag_b_kmajor(HT1 + j0 + ks, n, lane);
      c0 = __builtin_amdgcn_wmma_f32_16x16x32_f16(false, a0, false, b0, (short)0, c0, false, false);
      c1 = __builtin_amdgcn_wmma_f32_16x16x32_f16(false, a1, false, b1, (short)0, c1, false, false);
    }
    buf ^= 1;   // double buffer: next phase1 writes other buffer while peers may still read this one
  }
  // store partial accumulators: part[chunk][k][i][f]
  int nn = lane & 15, g = lane >> 4;
  float* P0 = part + ((size_t)chunk * 2 + 0) * n * C_DIM;
  float* P1 = part + ((size_t)chunk * 2 + 1) * n * C_DIM;
#pragma unroll
  for (int v = 0; v < 8; ++v) {
    size_t row = (size_t)(i0 + g * 8 + v);
    P0[row * C_DIM + n0 + nn] = c0[v];
    P1[row * C_DIM + n0 + nn] = c1[v];
  }
}

// sum chunk partials + residual + LN2 -> f16 y
__global__ void k_ln2(const float* __restrict__ part, const float* __restrict__ x,
                      const float* __restrict__ g2, const float* __restrict__ b2,
                      _Float16* __restrict__ y16, int n) {
  __shared__ float red[C_DIM];
  int idx = blockIdx.x;
  int k = idx / n, r = idx % n, t = threadIdx.x;
  float v = 0.f;
  for (int ch = 0; ch < NCHUNK; ++ch)
    v += part[((size_t)ch * 2 + k) * n * C_DIM + (size_t)r * C_DIM + t];
  v += x[(size_t)r * C_DIM + t];
  float mu = block_reduce_sum(v, red, t, C_DIM) * (1.f / C_DIM);
  float d = v - mu;
  float var = block_reduce_sum(d * d, red, t, C_DIM) * (1.f / C_DIM);
  float y = d * rsqrtf(var + LN_EPS) * g2[t] + b2[t];
  y16[((size_t)k * n + r) * C_DIM + t] = (_Float16)y;
}

// h_next = 0.5*(elu(y0@W1_0) + elu(y1@W1_1))
__global__ void k_gemm_out(const _Float16* __restrict__ y16, const _Float16* __restrict__ w1t,
                           float* __restrict__ out, int n) {
  int lane = threadIdx.x & 31;
  int wg = blockIdx.x * (blockDim.x >> 5) + (threadIdx.x >> 5);
  int tiles = (n / 16) * (C_DIM / 16);
  if (wg >= tiles) return;
  int mt = wg / (C_DIM / 16), nt = wg % (C_DIM / 16);
  v8f c0 = {}, c1 = {};
  const _Float16* A0 = y16 + (size_t)(mt * 16) * C_DIM;
  const _Float16* A1 = y16 + (size_t)n * C_DIM + (size_t)(mt * 16) * C_DIM;
  const _Float16* B0 = w1t + (size_t)(nt * 16) * C_DIM;
  const _Float16* B1 = w1t + (size_t)C_DIM * C_DIM + (size_t)(nt * 16) * C_DIM;
#pragma unroll
  for (int kk = 0; kk < C_DIM; kk += 32) {
    v16h a0 = load_frag_a(A0 + kk, C_DIM, lane);
    v16h a1 = load_frag_a(A1 + kk, C_DIM, lane);
    v16h b0 = load_frag_b_kmajor(B0 + kk, C_DIM, lane);
    v16h b1 = load_frag_b_kmajor(B1 + kk, C_DIM, lane);
    c0 = __builtin_amdgcn_wmma_f32_16x16x32_f16(false, a0, false, b0, (short)0, c0, false, false);
    c1 = __builtin_amdgcn_wmma_f32_16x16x32_f16(false, a1, false, b1, (short)0, c1, false, false);
  }
  int nn = lane & 15, g = lane >> 4;
#pragma unroll
  for (int v = 0; v < 8; ++v) {
    float g0 = c0[v], g1 = c1[v];
    g0 = g0 > 0.f ? g0 : (__expf(g0) - 1.f);
    g1 = g1 > 0.f ? g1 : (__expf(g1) - 1.f);
    out[(size_t)(mt * 16 + g * 8 + v) * C_DIM + nt * 16 + nn] = 0.5f * (g0 + g1);
  }
}

// -------------------- launch --------------------
extern "C" void kernel_launch(void* const* d_in, const int* in_sizes, int n_in,
                              void* d_out, int out_size, void* d_ws, size_t ws_size,
                              hipStream_t stream) {
  const float* x     = (const float*)d_in[0];
  const float* adj   = (const float*)d_in[1];
  const float* W0    = (const float*)d_in[2];
  const float* W1    = (const float*)d_in[3];
  const float* a     = (const float*)d_in[4];
  const float* ln1_g = (const float*)d_in[5];
  const float* ln1_b = (const float*)d_in[6];
  const float* ln2_g = (const float*)d_in[7];
  const float* ln2_b = (const float*)d_in[8];

  const int n = in_sizes[0] / C_DIM;   // 8192
  const int w0cnt = in_sizes[2];       // K*C*C
  const int w1cnt = in_sizes[3];

  float* h_next    = (float*)d_out;                       // n*C
  float* atten_agg = (float*)d_out + (size_t)n * C_DIM;   // n*n

  auto align = [](size_t v) { return (v + 255) & ~(size_t)255; };
  char* ws = (char*)d_ws;
  size_t off = 0;
  _Float16* xn16 = (_Float16*)(ws + off); off = align(off + (size_t)n * C_DIM * 2);
  _Float16* w0t  = (_Float16*)(ws + off); off = align(off + (size_t)w0cnt * 2);
  _Float16* w1t  = (_Float16*)(ws + off); off = align(off + (size_t)w1cnt * 2);
  _Float16* h16  = (_Float16*)(ws + off); off = align(off + (size_t)2 * n * C_DIM * 2);
  _Float16* ht16 = (_Float16*)(ws + off); off = align(off + (size_t)2 * n * C_DIM * 2);
  _Float16* y16  = (_Float16*)(ws + off); off = align(off + (size_t)2 * n * C_DIM * 2);
  float* er   = (float*)(ws + off); off = align(off + (size_t)2 * n * 4);
  float* ec   = (float*)(ws + off); off = align(off + (size_t)2 * n * 4);
  float* mrow = (float*)(ws + off); off = align(off + (size_t)2 * n * 4);
  float* sinv = (float*)(ws + off); off = align(off + (size_t)2 * n * 4);
  float* part = (float*)(ws + off); off = align(off + (size_t)NCHUNK * 2 * n * C_DIM * 4);
  (void)ws_size; (void)n_in; (void)out_size;

  k_ln1<<<n, C_DIM, 0, stream>>>(x, ln1_g, ln1_b, xn16);
  k_cvt_t<<<(w0cnt + 255) / 256, 256, 0, stream>>>(W0, w0t, w0cnt);
  k_cvt_t<<<(w1cnt + 255) / 256, 256, 0, stream>>>(W1, w1t, w1cnt);

  int tiles_h = 2 * (n / 16) * (C_DIM / 16);
  k_gemm_h<<<(tiles_h + 7) / 8, 256, 0, stream>>>(xn16, w0t, h16, ht16, n);

  k_scores<<<n, C_DIM, 0, stream>>>(h16, a, er, ec, n);
  k_stats<<<n, 256, 0, stream>>>(adj, er, ec, mrow, sinv, n);

  dim3 attn_grid(n / 16, NCHUNK);
  k_attn<<<attn_grid, 256, 0, stream>>>(adj, er, ec, mrow, sinv, ht16, atten_agg, part, n);

  k_ln2<<<2 * n, C_DIM, 0, stream>>>(part, x, ln2_g, ln2_b, y16, n);

  int tiles_o = (n / 16) * (C_DIM / 16);
  k_gemm_out<<<(tiles_o + 7) / 8, 256, 0, stream>>>(y16, w1t, h_next, n);
}
